// MultiHeadSelfAttention_42812234006591
// MI455X (gfx1250) — matrix operand order repaired and checked by reading, not compiled, not run on hardware
//
#include <hip/hip_runtime.h>
#include <math.h>

typedef __attribute__((ext_vector_type(16))) _Float16 v16h;
typedef __attribute__((ext_vector_type(8)))  _Float16 v8h;
typedef __attribute__((ext_vector_type(8)))  float    v8f;
typedef __attribute__((ext_vector_type(4)))  unsigned u32x4;
typedef __attribute__((ext_vector_type(8)))  int      i32x8;
typedef __attribute__((ext_vector_type(4)))  int      i32x4;

#define DD   768
#define SS   4096
#define BB   2
#define NH   12
#define HD   64
#define BSR  (BB*SS)   // 8192 rows

#if __has_builtin(__builtin_amdgcn_tensor_load_to_lds) && \
    __has_builtin(__builtin_amdgcn_s_wait_tensorcnt)
#define USE_TDM 1
#else
#define USE_TDM 0
#endif

// ---------------------------------------------------------------------------
// A-fragment (16x32 f16) gather in contiguous k order (not the ISA table order; B loads match):
// lane L: m = L%16, h = L/16; per h the halfs are contiguous 32B
// at k = 16h..16h+15  -> lowers to 2x ds_load_b128.
// ---------------------------------------------------------------------------
__device__ __forceinline__ v16h load_a_frag(const _Float16* __restrict__ row, int h) {
  union { v16h v; _Float16 e[16]; } u;
#pragma unroll
  for (int p = 0; p < 8; ++p) {
    int k0 = 16 * h + 2 * p;
    u.e[2 * p]     = row[k0];
    u.e[2 * p + 1] = row[k0 + 1];
  }
  return u.v;
}

__device__ __forceinline__ v8f wmma32(v16h a, v16h b, v8f c) {
  return __builtin_amdgcn_wmma_f32_16x16x32_f16(false, a, false, b, (short)0, c,
                                                false, false);
}

#if USE_TDM
__device__ __forceinline__ unsigned lds_addr_of(const void* p) {
  return (unsigned)(size_t)(__attribute__((address_space(3))) const char*)p;
}

// 2-D tile DMA: global (row-major, stride0 elements of 2B) -> contiguous LDS.
// D# packing per CDNA5 ISA §8.3 (group0) / §8.4 (group1); groups 2/3 zero (2D).
// This toolchain exposes the 6-arg builtin:
//   (u32x4 g0, i32x8 g1, i32x4, i32x4, i32x8, i32 cpol)
__device__ __forceinline__ void tdm_load_2d(unsigned lds_off, const void* gp,
                                            unsigned tile_d0, unsigned tile_d1,
                                            unsigned tensor_d0, unsigned tensor_d1,
                                            unsigned stride0) {
  unsigned long long ga = (unsigned long long)gp;
  u32x4 g0;
  g0[0] = 1u;                                   // count=1, user mode
  g0[1] = lds_off;                              // lds_addr (bytes)
  g0[2] = (unsigned)(ga & 0xFFFFFFFFu);         // global_addr[31:0]
  g0[3] = (unsigned)((ga >> 32) & 0x1FFFFFFu)   // global_addr[56:32]
          | (2u << 30);                         // type = 2 ("image")
  i32x8 g1;
  g1[0] = (int)(1u << 16);                      // wg_mask=0, data_size=1 (2B)
  g1[1] = (int)(tensor_d0 << 16);               // [63:48]=tensor_dim0[15:0]
  g1[2] = (int)((tensor_d0 >> 16) | (tensor_d1 << 16));
  g1[3] = (int)((tensor_d1 >> 16) | (tile_d0 << 16));
  g1[4] = (int)tile_d1;                         // tile_dim1 ([143:128]), tile_dim2=0
  g1[5] = (int)stride0;                         // tensor_dim0_stride[31:0]
  g1[6] = 0;
  g1[7] = 0;
  i32x4 z4 = {0, 0, 0, 0};
  i32x8 z8 = {0, 0, 0, 0, 0, 0, 0, 0};
  __builtin_amdgcn_tensor_load_to_lds(g0, g1, z4, z4, z8, 0);
}
#endif

// ---------------------------------------------------------------------------
// Weight convert+transpose: W[in][out] f32 -> Wt[out][in] f16
// ---------------------------------------------------------------------------
__global__ __launch_bounds__(256) void wt_kernel(const float* __restrict__ W,
                                                 _Float16* __restrict__ Wt) {
  int idx = blockIdx.x * 256 + threadIdx.x;       // 768*768 elems
  int i = idx / DD;   // in
  int o = idx % DD;   // out
  Wt[(size_t)o * DD + i] = (_Float16)W[idx];
}

// ---------------------------------------------------------------------------
// LayerNorm: one block per row of x[8192][768], writes f16 xn
// ---------------------------------------------------------------------------
__global__ __launch_bounds__(256) void ln_kernel(const float* __restrict__ x,
                                                 const float* __restrict__ g,
                                                 const float* __restrict__ b,
                                                 _Float16* __restrict__ xn16) {
  __shared__ float s1[256];
  __shared__ float s2[256];
  const int row = blockIdx.x;
  const int t   = threadIdx.x;
  const float* xr = x + (size_t)row * DD;
  float v0 = xr[t], v1 = xr[t + 256], v2 = xr[t + 512];
  s1[t] = v0 + v1 + v2;
  s2[t] = v0 * v0 + v1 * v1 + v2 * v2;
  __syncthreads();
  for (int off = 128; off > 0; off >>= 1) {
    if (t < off) { s1[t] += s1[t + off]; s2[t] += s2[t + off]; }
    __syncthreads();
  }
  float mu  = s1[0] * (1.0f / DD);
  float var = s2[0] * (1.0f / DD) - mu * mu;
  float rs  = rsqrtf(var + 1e-5f);
  _Float16* xo = xn16 + (size_t)row * DD;
  xo[t]       = (_Float16)((v0 - mu) * rs * g[t]       + b[t]);
  xo[t + 256] = (_Float16)((v1 - mu) * rs * g[t + 256] + b[t + 256]);
  xo[t + 512] = (_Float16)((v2 - mu) * rs * g[t + 512] + b[t + 512]);
}

// ---------------------------------------------------------------------------
// GEMM: C[m][n] = A[m][:] . Wt[n][:] + bias[n]  (M=8192, N=768, K=768)
// BM=256, BN=64, BK=32; 256 threads = 8 waves; wave w -> rows [32w, 32w+32).
// Per k-step each wave: 2 A-frags + 4 B-frags -> 8 WMMAs (B reused across M).
// mode 0: out f16 at [(b*NH+head)*S + s]*64 + hd      (Q / K layout)
// mode 1: out f16 at [(b*NH+head)*64 + hd]*S + s      (V transposed)
// mode 2: out f32 = acc + bias + residual             (final projection)
// ---------------------------------------------------------------------------
__global__ __launch_bounds__(256) void gemm_kernel(
    const _Float16* __restrict__ A, const _Float16* __restrict__ Wt,
    const float* __restrict__ bias, const float* __restrict__ residual,
    _Float16* __restrict__ out16, float* __restrict__ out32, int mode) {
  __shared__ alignas(32) _Float16 As[256 * 32];   // 16 KB
  __shared__ alignas(32) _Float16 Bs[64 * 32];    //  4 KB
  const int m0   = blockIdx.x * 256;
  const int n0   = blockIdx.y * 64;
  const int tid  = threadIdx.x;
  const int wave = tid >> 5;
  const int lane = tid & 31;
  const int h    = lane >> 4;
  const int ln   = lane & 15;

  v8f acc[2][4];
#pragma unroll
  for (int i = 0; i < 2; ++i)
#pragma unroll
    for (int j = 0; j < 4; ++j) acc[i][j] = (v8f){0, 0, 0, 0, 0, 0, 0, 0};

#if USE_TDM
  const unsigned As_off = lds_addr_of(As);
  const unsigned Bs_off = lds_addr_of(Bs);
#endif

  for (int k0 = 0; k0 < DD; k0 += 32) {
#if USE_TDM
    if (wave == 0) {
      tdm_load_2d(As_off, &A[(size_t)m0 * DD + k0], 32, 256, DD, BSR, DD);
      tdm_load_2d(Bs_off, &Wt[(size_t)n0 * DD + k0], 32, 64, DD, DD, DD);
      __builtin_amdgcn_s_wait_tensorcnt(0);
    }
    __syncthreads();
#else
    {  // stage A tile 256x32: one row (32 halfs) per thread
      *(v16h*)&As[tid * 32] = *(const v16h*)&A[(size_t)(m0 + tid) * DD + k0];
      *(v16h*)&As[tid * 32 + 16] =
          *(const v16h*)&A[(size_t)(m0 + tid) * DD + k0 + 16];
    }
    {  // stage B tile 64x32
      int row = tid >> 2, part = tid & 3;
      *(v8h*)&Bs[row * 32 + part * 8] =
          *(const v8h*)&Wt[(size_t)(n0 + row) * DD + k0 + part * 8];
    }
    __syncthreads();
#endif
    v16h a0 = load_a_frag(&As[(wave * 32 + ln) * 32], h);
    v16h a1 = load_a_frag(&As[(wave * 32 + 16 + ln) * 32], h);
#pragma unroll
    for (int nf = 0; nf < 4; ++nf) {
      v16h b = *(const v16h*)&Bs[(nf * 16 + ln) * 32 + 16 * h];
      acc[0][nf] = wmma32(a0, b, acc[0][nf]);
      acc[1][nf] = wmma32(a1, b, acc[1][nf]);
    }
    __syncthreads();
  }

#pragma unroll
  for (int mi = 0; mi < 2; ++mi) {
#pragma unroll
    for (int nf = 0; nf < 4; ++nf) {
      int nn = n0 + nf * 16 + ln;
      float bv = bias[nn];
#pragma unroll
      for (int r = 0; r < 8; ++r) {
        int m = m0 + wave * 32 + mi * 16 + 8 * h + r;
        float val = acc[mi][nf][r] + bv;
        if (mode == 2) {
          out32[(size_t)m * DD + nn] = val + residual[(size_t)m * DD + nn];
        } else {
          int b_ = m / SS, s_ = m % SS;
          int head = nn / HD, hd = nn % HD;
          size_t idx = (mode == 0)
              ? (((size_t)(b_ * NH + head)) * SS + s_) * HD + hd
              : (((size_t)(b_ * NH + head)) * HD + hd) * SS + s_;
          out16[idx] = (_Float16)val;
        }
      }
    }
  }
}

// ---------------------------------------------------------------------------
// Flash attention: grid (32 q-tiles, 24 b*h). 8 waves x 16 query rows.
// Key tile = 64: 16 WMMAs per tile, softmax shuffles amortized over 64 keys.
// Q,K: [bh][S][64] f16; Vt: [bh][64][S] f16; O: [B][S][768] f16.
// ---------------------------------------------------------------------------
__global__ __launch_bounds__(256) void attn_kernel(
    const _Float16* __restrict__ Q, const _Float16* __restrict__ K,
    const _Float16* __restrict__ Vt, _Float16* __restrict__ O) {
  __shared__ alignas(32) _Float16 P[8 * 16 * 64];  // per-wave 16x64 P tile
  const int qt   = blockIdx.x;   // 0..31
  const int bh   = blockIdx.y;   // 0..23
  const int tid  = threadIdx.x;
  const int wave = tid >> 5;
  const int lane = tid & 31;
  const int h    = lane >> 4;
  const int ln   = lane & 15;
  const int q0   = qt * 128 + wave * 16;

  // Q fragments for this wave's 16 rows, pre-scaled by 1/sqrt(64)
  const _Float16* qrow = Q + ((size_t)bh * SS + q0 + ln) * HD;
  v16h qa0 = load_a_frag(qrow, h);
  v16h qa1 = load_a_frag(qrow + 32, h);
#pragma unroll
  for (int i = 0; i < 16; ++i) {
    qa0[i] = qa0[i] * (_Float16)0.125f;
    qa1[i] = qa1[i] * (_Float16)0.125f;
  }

  const v8f vz = (v8f){0, 0, 0, 0, 0, 0, 0, 0};
  v8f o[4] = {vz, vz, vz, vz};
  float mr[8], lr[8];
#pragma unroll
  for (int r = 0; r < 8; ++r) { mr[r] = -1e30f; lr[r] = 0.0f; }

  _Float16* Pw = &P[wave * 16 * 64];
  const _Float16* Kb = K + (size_t)bh * SS * HD;
  const _Float16* Vb = Vt + (size_t)bh * HD * SS;

  for (int j = 0; j < SS; j += 64) {
    // scores for 4 groups of 16 keys: s[kg], key = j + kg*16 + ln
    v8f s[4];
#pragma unroll
    for (int kg = 0; kg < 4; ++kg) {
      const _Float16* krow = &Kb[(size_t)(j + kg * 16 + ln) * HD + 16 * h];
      v16h kb_lo = *(const v16h*)krow;
      v16h kb_hi = *(const v16h*)(krow + 32);
      v8f t = vz;
      t = wmma32(qa0, kb_lo, t);
      t = wmma32(qa1, kb_hi, t);
      s[kg] = t;
    }

    // online softmax per query row (lane holds rows 8h+r, col n=ln)
#pragma unroll
    for (int r = 0; r < 8; ++r) {
      float a0 = s[0][r], a1 = s[1][r], a2 = s[2][r], a3 = s[3][r];
      float mx = fmaxf(fmaxf(a0, a1), fmaxf(a2, a3));
      mx = fmaxf(mx, __shfl_xor(mx, 1, 32));
      mx = fmaxf(mx, __shfl_xor(mx, 2, 32));
      mx = fmaxf(mx, __shfl_xor(mx, 4, 32));
      mx = fmaxf(mx, __shfl_xor(mx, 8, 32));
      float nm    = fmaxf(mr[r], mx);
      float alpha = __expf(mr[r] - nm);
      float p0 = __expf(a0 - nm);
      float p1 = __expf(a1 - nm);
      float p2 = __expf(a2 - nm);
      float p3 = __expf(a3 - nm);
      float ps = (p0 + p1) + (p2 + p3);
      ps += __shfl_xor(ps, 1, 32);
      ps += __shfl_xor(ps, 2, 32);
      ps += __shfl_xor(ps, 4, 32);
      ps += __shfl_xor(ps, 8, 32);
      lr[r] = lr[r] * alpha + ps;
      mr[r] = nm;
      o[0][r] *= alpha; o[1][r] *= alpha; o[2][r] *= alpha; o[3][r] *= alpha;
      _Float16* pr = &Pw[(8 * h + r) * 64 + ln];
      pr[0]  = (_Float16)p0;
      pr[16] = (_Float16)p1;
      pr[32] = (_Float16)p2;
      pr[48] = (_Float16)p3;
    }

    // P (16x64) as two A-fragments via LDS relayout; V^T contiguous in key
    v16h pa0 = load_a_frag(&Pw[ln * 64], h);
    v16h pa1 = load_a_frag(&Pw[ln * 64 + 32], h);
#pragma unroll
    for (int nf = 0; nf < 4; ++nf) {
      const _Float16* vrow = &Vb[(size_t)(nf * 16 + ln) * SS + j + 16 * h];
      v16h vb_lo = *(const v16h*)vrow;
      v16h vb_hi = *(const v16h*)(vrow + 32);
      o[nf] = wmma32(pa0, vb_lo, o[nf]);
      o[nf] = wmma32(pa1, vb_hi, o[nf]);
    }
  }

  const int b_   = bh / NH;
  const int head = bh % NH;
#pragma unroll
  for (int r = 0; r < 8; ++r) {
    float inv = 1.0f / lr[r];
    int srow = q0 + 8 * h + r;
    size_t base = ((size_t)b_ * SS + srow) * DD + head * HD;
    O[base + 0  + ln] = (_Float16)(o[0][r] * inv);
    O[base + 16 + ln] = (_Float16)(o[1][r] * inv);
    O[base + 32 + ln] = (_Float16)(o[2][r] * inv);
    O[base + 48 + ln] = (_Float16)(o[3][r] * inv);
  }
}

// ---------------------------------------------------------------------------
extern "C" void kernel_launch(void* const* d_in, const int* in_sizes, int n_in,
                              void* d_out, int out_size, void* d_ws,
                              size_t ws_size, hipStream_t stream) {
  const float* x    = (const float*)d_in[0];
  const float* Wq   = (const float*)d_in[1];
  const float* bq   = (const float*)d_in[2];
  const float* Wk   = (const float*)d_in[3];
  const float* bk   = (const float*)d_in[4];
  const float* Wv   = (const float*)d_in[5];
  const float* bv   = (const float*)d_in[6];
  const float* Wo   = (const float*)d_in[7];
  const float* bo   = (const float*)d_in[8];
  const float* ln_g = (const float*)d_in[9];
  const float* ln_b = (const float*)d_in[10];
  float* out = (float*)d_out;

  _Float16* ws = (_Float16*)d_ws;
  size_t off = 0;
  _Float16* xn16 = ws + off; off += (size_t)BSR * DD;
  _Float16* Wqt  = ws + off; off += (size_t)DD * DD;
  _Float16* Wkt  = ws + off; off += (size_t)DD * DD;
  _Float16* Wvt  = ws + off; off += (size_t)DD * DD;
  _Float16* Wot  = ws + off; off += (size_t)DD * DD;
  _Float16* Qb   = ws + off; off += (size_t)BSR * DD;
  _Float16* Kbuf = ws + off; off += (size_t)BSR * DD;
  _Float16* Vtb  = ws + off; off += (size_t)BSR * DD;
  _Float16* attn = xn16;  // xn16 dead after V GEMM; same-stream reuse is safe

  const int wgrid = (DD * DD) / 256;  // 2304
  wt_kernel<<<wgrid, 256, 0, stream>>>(Wq, Wqt);
  wt_kernel<<<wgrid, 256, 0, stream>>>(Wk, Wkt);
  wt_kernel<<<wgrid, 256, 0, stream>>>(Wv, Wvt);
  wt_kernel<<<wgrid, 256, 0, stream>>>(Wo, Wot);

  ln_kernel<<<BSR, 256, 0, stream>>>(x, ln_g, ln_b, xn16);

  dim3 gg(BSR / 256, DD / 64);  // 32 x 12
  gemm_kernel<<<gg, 256, 0, stream>>>(xn16, Wqt, bq, nullptr, Qb,   nullptr, 0);
  gemm_kernel<<<gg, 256, 0, stream>>>(xn16, Wkt, bk, nullptr, Kbuf, nullptr, 0);
  gemm_kernel<<<gg, 256, 0, stream>>>(xn16, Wvt, bv, nullptr, Vtb,  nullptr, 1);

  attn_kernel<<<dim3(SS / 128, BB * NH), 256, 0, stream>>>(Qb, Kbuf, Vtb, attn);

  gemm_kernel<<<gg, 256, 0, stream>>>(attn, Wot, bo, x, nullptr, out, 2);
}
